// SelfAttention_678604832919
// MI455X (gfx1250) — compile-verified, matrix operand order checked by reading, not run on hardware
//
#include <hip/hip_runtime.h>

// ---------------------------------------------------------------------------
// SAGAN self-attention w/ spectral norm, CDNA5 (gfx1250) wave32 WMMA version.
// Pipeline: sigma_kernel -> qkv_kernel (bf16 WMMA GEMM) -> attn_kernel
// (flash-style online softmax over the query axis + bf16 WMMA PV GEMM).
// ---------------------------------------------------------------------------

typedef __attribute__((ext_vector_type(16))) __bf16 v16bf;
typedef __attribute__((ext_vector_type(8)))  __bf16 v8bf;
typedef __attribute__((ext_vector_type(8)))  float  v8f;

#define BATCH 32
#define CCH   512
#define CQ    64
#define NSP   1024   // 32*32 spatial

// LDS row padding (in bf16 elements) to de-conflict banks for b128 reads.
#define WPAD 72      // 64 + 8  -> 144B stride = 36 dwords (permutes banks)
#define PPAD 136     // 128 + 8 -> 272B stride = 68 dwords (lane m -> bank 4m)

// 16-bit A/B operand striping (ISA 7.12.2): lane (half = lane>>4) owns
// K = half*8 + {0..7} and 16 + half*8 + {0..7} -> two contiguous 16B chunks.
__device__ __forceinline__ v16bf load_ab(const __bf16* row, int half) {
  v8bf lo = *(const v8bf*)(row + half * 8);
  v8bf hi = *(const v8bf*)(row + 16 + half * 8);
  v16bf r;
#pragma unroll
  for (int i = 0; i < 8; ++i) { r[i] = lo[i]; r[i + 8] = hi[i]; }
  return r;
}

__device__ __forceinline__ v8f wmma_bf16(v16bf a, v16bf b, v8f c) {
  return __builtin_amdgcn_wmma_f32_16x16x32_bf16(
      false, a, false, b, (short)0, c, false, false);
}

// ---------------------------------------------------------------------------
// Kernel 1: scale = 1/sigma, sigma = || W @ normalize(W^T u) ||.
// blockIdx.x selects {Wq, Wk, Wv}. 3 blocks x 256 threads; trivial cost.
// ---------------------------------------------------------------------------
__global__ __launch_bounds__(256) void sigma_kernel(
    const float* __restrict__ Wq, const float* __restrict__ uq,
    const float* __restrict__ Wk, const float* __restrict__ uk,
    const float* __restrict__ Wv, const float* __restrict__ uv,
    float* __restrict__ scales) {
  __shared__ float red[256];
  __shared__ float vvec[CCH];
  const int which = blockIdx.x;
  const float* W = (which == 0) ? Wq : (which == 1) ? Wk : Wv;
  const float* u = (which == 0) ? uq : (which == 1) ? uk : uv;
  const int outd = (which == 2) ? CCH : CQ;
  const int tid = threadIdx.x;

  // t = W^T u ; accumulate ||t||^2
  float nrm2 = 0.f;
  for (int i = tid; i < CCH; i += 256) {
    float acc = 0.f;
    for (int o = 0; o < outd; ++o) acc += W[o * CCH + i] * u[o];
    vvec[i] = acc;
    nrm2 += acc * acc;
  }
  red[tid] = nrm2;
  __syncthreads();
  for (int s = 128; s > 0; s >>= 1) {
    if (tid < s) red[tid] += red[tid + s];
    __syncthreads();
  }
  const float inv = rsqrtf(red[0] + 1e-30f);  // 1/||t||
  __syncthreads();

  // w = W (t/||t||) ; sigma^2 = ||w||^2
  float sig2 = 0.f;
  for (int o = tid; o < outd; o += 256) {
    float acc = 0.f;
    for (int i = 0; i < CCH; ++i) acc += W[o * CCH + i] * vvec[i];
    acc *= inv;
    sig2 += acc * acc;
  }
  red[tid] = sig2;
  __syncthreads();
  for (int s = 128; s > 0; s >>= 1) {
    if (tid < s) red[tid] += red[tid + s];
    __syncthreads();
  }
  if (tid == 0) scales[which] = rsqrtf(red[0] + 1e-30f);  // 1/sigma
}

// ---------------------------------------------------------------------------
// Kernel 2: packed QKV projection.  out[o, n] = (W_sn @ f)[o, n] + bias,
// o in [0,640): rows 0..63 -> q, 64..127 -> k, 128..639 -> v.
// Grid (N/32, 640/64, B), block 256 = 8 waves; wave (w&3,w>>2) owns a 16x16
// subtile of the 64x32 block tile.  64-wide K-chunks staged in LDS (scaled
// bf16 W, transposed bf16 f) -> 2 WMMAs per barrier pair, 8 K-steps.
//   qbuf[b][n][cq]  (bf16)   kbuf[b][m][cq] (bf16)   vbuf[b][c][n] (bf16)
// ---------------------------------------------------------------------------
__global__ __launch_bounds__(256) void qkv_kernel(
    const float* __restrict__ x,
    const float* __restrict__ Wq, const float* __restrict__ bq,
    const float* __restrict__ Wk, const float* __restrict__ bk,
    const float* __restrict__ Wv, const float* __restrict__ bv,
    const float* __restrict__ scales,
    __bf16* __restrict__ qbuf, __bf16* __restrict__ kbuf,
    __bf16* __restrict__ vbuf) {
  __shared__ __bf16 Wt[64][WPAD];  // [o_local][c_local], pre-scaled bf16
  __shared__ __bf16 Ft[32][WPAD];  // [n_local][c_local] (transposed stage)

  const int b  = blockIdx.z;
  const int n0 = blockIdx.x * 32;
  const int o0 = blockIdx.y * 64;
  const int tid  = threadIdx.x;
  const int w    = tid >> 5;
  const int lane = tid & 31;
  const int half = lane >> 4;
  const int lm   = lane & 15;
  const int sub_o = (w & 3) * 16;
  const int sub_n = (w >> 2) * 16;

  // region select (uniform per block; o-tiles of 64 never straddle regions)
  const float* Wsrc; const float* bsrc; float sc;
  if (o0 < 64)        { Wsrc = Wq; bsrc = bq; sc = scales[0]; }
  else if (o0 < 128)  { Wsrc = Wk; bsrc = bk; sc = scales[1]; }
  else                { Wsrc = Wv; bsrc = bv; sc = scales[2]; }
  const int obase = (o0 < 64) ? 0 : (o0 < 128) ? 64 : 128;
  const int orow0 = o0 - obase;

  v8f acc = {0.f, 0.f, 0.f, 0.f, 0.f, 0.f, 0.f, 0.f};

  for (int c0 = 0; c0 < CCH; c0 += 64) {
    __syncthreads();  // LDS reuse fence
    // stage W tile (64x64), scaled + converted to bf16; coalesced over c
#pragma unroll
    for (int kk = 0; kk < 16; ++kk) {
      const int e = tid + kk * 256;
      const int r = e >> 6, c = e & 63;
      Wt[r][c] = (__bf16)(Wsrc[(size_t)(orow0 + r) * CCH + c0 + c] * sc);
    }
    // stage f tile transposed: coalesced read over n, write [n][c]
#pragma unroll
    for (int kk = 0; kk < 8; ++kk) {
      const int e = tid + kk * 256;
      const int cc = e >> 5, nn = e & 31;
      const float fv = x[((size_t)b * CCH + (c0 + cc)) * NSP + n0 + nn];
      Ft[nn][cc] = (__bf16)fv;
    }
    // prefetch next K-chunk while this one is consumed
    if (c0 + 64 < CCH) {
      __builtin_prefetch(
          &Wsrc[(size_t)(orow0 + (tid >> 2)) * CCH + c0 + 64 + (tid & 3) * 16],
          0, 3);
      __builtin_prefetch(
          &x[((size_t)b * CCH + (c0 + 64 + (tid >> 2))) * NSP + n0 + (tid & 3) * 8],
          0, 3);
    }
    __syncthreads();
    const __bf16* wr = &Wt[sub_o + lm][0];
    const __bf16* fr = &Ft[sub_n + lm][0];
    acc = wmma_bf16(load_ab(wr, half),      load_ab(fr, half),      acc);
    acc = wmma_bf16(load_ab(wr + 32, half), load_ab(fr + 32, half), acc);
  }

  // epilogue: C/D layout lane(L) holds row r + 8*half, col (L&15).
  // Within a lane the 8 rows are CONTIGUOUS channels -> pack b128 for q/k.
  const int n_g = n0 + sub_n + lm;
  if (o0 < 128) {
    // q (o0==0) or k (o0==64): [n][cq] layout, 8 contiguous cq per lane
    const int cbase = sub_o + 8 * half;  // channel index within region
    v8bf pk;
#pragma unroll
    for (int r = 0; r < 8; ++r) pk[r] = (__bf16)(acc[r] + bsrc[cbase + r]);
    __bf16* dst = (o0 < 64) ? qbuf : kbuf;
    *(v8bf*)(dst + ((size_t)b * NSP + n_g) * CQ + cbase) = pk;
  } else {
    // v: [c][n] layout; rows strided by NSP, one b16 store per row
#pragma unroll
    for (int r = 0; r < 8; ++r) {
      const int crow = orow0 + sub_o + r + 8 * half;
      vbuf[((size_t)b * CCH + crow) * NSP + n_g] =
          (__bf16)(acc[r] + bsrc[crow]);
    }
  }
}

// ---------------------------------------------------------------------------
// Kernel 3: flash-style attention with softmax over n (the contraction axis
// of v@attn).  Grid (N/16, B); block 128 = 4 waves.  Wave w owns C-rows
// [w*128, w*128+128) -> 8 f32 accumulator tiles; per 128-wide n-step, wave w
// computes its own 32x16 score chunk (S = q(n) . k(m), K=64), cross-wave
// max/sum via LDS, P = exp(S - M) staged transposed as bf16 in LDS, then
// O += v @ P (4 K-chunks x 8 c-subtiles = 32 WMMAs per wave per step).
// ---------------------------------------------------------------------------
__global__ __launch_bounds__(128) void attn_kernel(
    const float* __restrict__ x,
    const __bf16* __restrict__ qbuf, const __bf16* __restrict__ kbuf,
    const __bf16* __restrict__ vbuf, const float* __restrict__ gamma_p,
    float* __restrict__ out) {
  __shared__ float  colmax[4 * 16];
  __shared__ float  colsum[4 * 16];
  __shared__ __bf16 Pt[16][PPAD];  // [m_local][n_local], bank-conflict free

  const int b  = blockIdx.y;
  const int m0 = blockIdx.x * 16;
  const int tid  = threadIdx.x;
  const int w    = tid >> 5;
  const int lane = tid & 31;
  const int half = lane >> 4;
  const int lm   = lane & 15;
  const float gamma = gamma_p[0];

  const __bf16* qb = qbuf + (size_t)b * NSP * CQ;
  const __bf16* kb = kbuf + (size_t)b * NSP * CQ;
  const __bf16* vb = vbuf + (size_t)b * CCH * NSP;

  float Mrun = -1e30f, Lrun = 0.f;
  v8f Oacc[8];
#pragma unroll
  for (int t = 0; t < 8; ++t) Oacc[t] = (v8f){0.f,0.f,0.f,0.f,0.f,0.f,0.f,0.f};

  // k columns for this m-tile are fixed: load B operands once (K=64 -> 2)
  const __bf16* krow = kb + (size_t)(m0 + lm) * CQ;
  const v16bf Bk0 = load_ab(krow, half);
  const v16bf Bk1 = load_ab(krow + 32, half);

  for (int n0 = 0; n0 < NSP; n0 += 128) {
    const int nw = n0 + w * 32;  // this wave's private 32-row score chunk

    // prefetch next n-chunk of v while computing this one
    if (n0 + 128 < NSP) {
#pragma unroll
      for (int t = 0; t < 8; ++t)
        __builtin_prefetch(
            vb + (size_t)(w * 128 + t * 16 + lm) * NSP + n0 + 128 + half * 64,
            0, 3);
    }

    // ---- scores: S[nw..nw+32, m0..m0+16] ----
    v8f s0 = {0.f,0.f,0.f,0.f,0.f,0.f,0.f,0.f};
    v8f s1 = {0.f,0.f,0.f,0.f,0.f,0.f,0.f,0.f};
    {
      const __bf16* qr0 = qb + (size_t)(nw + lm) * CQ;
      s0 = wmma_bf16(load_ab(qr0, half),      Bk0, s0);
      s0 = wmma_bf16(load_ab(qr0 + 32, half), Bk1, s0);
      const __bf16* qr1 = qb + (size_t)(nw + 16 + lm) * CQ;
      s1 = wmma_bf16(load_ab(qr1, half),      Bk0, s1);
      s1 = wmma_bf16(load_ab(qr1 + 32, half), Bk1, s1);
    }

    // ---- column max over this 32-row chunk (column m = lm) ----
    float mx = s0[0];
#pragma unroll
    for (int r = 0; r < 8; ++r) { mx = fmaxf(mx, s0[r]); mx = fmaxf(mx, s1[r]); }
    mx = fmaxf(mx, __shfl_xor(mx, 16, 32));
    if (half == 0) colmax[w * 16 + lm] = mx;
    __syncthreads();
    const float chm = fmaxf(fmaxf(colmax[lm], colmax[16 + lm]),
                            fmaxf(colmax[32 + lm], colmax[48 + lm]));
    const float Mnew  = fmaxf(Mrun, chm);
    const float alpha = __expf(Mrun - Mnew);

    // ---- P = exp(S - M), partial sums, stage transposed bf16 to LDS ----
    float psum = 0.f;
    __bf16* prow = &Pt[lm][w * 32];
#pragma unroll
    for (int r = 0; r < 8; ++r) {
      const float p0 = __expf(s0[r] - Mnew);
      const float p1 = __expf(s1[r] - Mnew);
      psum += p0 + p1;
      prow[r + 8 * half]      = (__bf16)p0;
      prow[16 + r + 8 * half] = (__bf16)p1;
    }
    psum += __shfl_xor(psum, 16, 32);
    if (half == 0) colsum[w * 16 + lm] = psum;
    __syncthreads();
    const float chs =
        colsum[lm] + colsum[16 + lm] + colsum[32 + lm] + colsum[48 + lm];
    Lrun = Lrun * alpha + chs;
    Mrun = Mnew;
#pragma unroll
    for (int t = 0; t < 8; ++t) {
#pragma unroll
      for (int r = 0; r < 8; ++r) Oacc[t][r] *= alpha;
    }

    // ---- O += v[c, n-chunk] @ P[n-chunk, m] ----
#pragma unroll
    for (int nn = 0; nn < 128; nn += 32) {
      const v16bf Bp = load_ab(&Pt[lm][nn], half);
#pragma unroll
      for (int t = 0; t < 8; ++t) {
        const int crow = w * 128 + t * 16 + lm;
        const __bf16* vrow = vb + (size_t)crow * NSP + n0 + nn;
        Oacc[t] = wmma_bf16(load_ab(vrow, half), Bp, Oacc[t]);
      }
    }
    __syncthreads();  // protect colmax/colsum/Pt for next n-step
  }

  // ---- epilogue: normalize, gamma*o + x ----
  const float rinv = 1.0f / Lrun;
  const int mg = m0 + lm;
#pragma unroll
  for (int t = 0; t < 8; ++t) {
#pragma unroll
    for (int r = 0; r < 8; ++r) {
      const int c = w * 128 + t * 16 + r + 8 * half;
      const size_t idx = ((size_t)b * CCH + c) * NSP + mg;
      out[idx] = gamma * (Oacc[t][r] * rinv) + x[idx];
    }
  }
}

// ---------------------------------------------------------------------------
extern "C" void kernel_launch(void* const* d_in, const int* in_sizes, int n_in,
                              void* d_out, int out_size, void* d_ws,
                              size_t ws_size, hipStream_t stream) {
  (void)in_sizes; (void)n_in; (void)out_size; (void)ws_size;
  const float* x  = (const float*)d_in[0];
  const float* Wq = (const float*)d_in[1];
  const float* bq = (const float*)d_in[2];
  const float* uq = (const float*)d_in[3];
  const float* Wk = (const float*)d_in[4];
  const float* bk = (const float*)d_in[5];
  const float* uk = (const float*)d_in[6];
  const float* Wv = (const float*)d_in[7];
  const float* bv = (const float*)d_in[8];
  const float* uv = (const float*)d_in[9];
  const float* gm = (const float*)d_in[10];
  float* out = (float*)d_out;

  // workspace layout (all 256B aligned):
  //   [0..12)           : scales (1/sigma for q,k,v)
  //   [256  .. +4MB)    : qbuf  bf16 [B][N][Cq]
  //   [+4MB .. +8MB)    : kbuf  bf16 [B][M][Cq]
  //   [+8MB .. +40MB)   : vbuf  bf16 [B][C][N]
  char* ws = (char*)d_ws;
  float* scales = (float*)ws;
  const size_t QK_BYTES = (size_t)BATCH * NSP * CQ * sizeof(__bf16);  // 4 MB
  __bf16* qbuf = (__bf16*)(ws + 256);
  __bf16* kbuf = (__bf16*)(ws + 256 + QK_BYTES);
  __bf16* vbuf = (__bf16*)(ws + 256 + 2 * QK_BYTES);

  sigma_kernel<<<3, 256, 0, stream>>>(Wq, uq, Wk, uk, Wv, uv, scales);

  qkv_kernel<<<dim3(NSP / 32, 640 / 64, BATCH), 256, 0, stream>>>(
      x, Wq, bq, Wk, bk, Wv, bv, scales, qbuf, kbuf, vbuf);

  attn_kernel<<<dim3(NSP / 16, BATCH), 128, 0, stream>>>(
      x, qbuf, kbuf, vbuf, gm, out);
}
